// Att_mlp_softmax_88545045774671
// MI455X (gfx1250) — matrix operand order repaired and checked by reading, not compiled, not run on hardware
//
#include <hip/hip_runtime.h>
#include <cstdint>
#include <cmath>

// ---------------------------------------------------------------------------
// Problem constants (from the reference)
// ---------------------------------------------------------------------------
#define N_NODES 8192
#define DIMS    128
#define HID     64
#define NCOLS   144            // 128 h-cols + w-col + ones-col + 14 zero pad
#define NT      9              // NCOLS / 16
#define KSPLIT  4
#define KCHUNK  (N_NODES / KSPLIT)   // 2048
#define MTILE   128            // rows per block (8 waves x 16 rows)

// Workspace layout (bytes)
#define WS_E    0u             // e[8192]  f32              (32 KB)
#define WS_GMAX 32768u         // gmax     f32
#define WS_B    65536u         // B fragments bf16: 256 ktiles * 9 ntiles * 512 halfs = 2.25 MB
#define WS_SN   (65536u + 2359296u)  // sn[8192][16] f32 (s at col0, num at col1) 512 KB

typedef __attribute__((ext_vector_type(16))) __bf16    v16bf;
typedef __attribute__((ext_vector_type(8)))  float     v8f;
typedef __attribute__((ext_vector_type(8)))  uint32_t  v8u;

// v_perm_b32: build one dword holding {bf16(hi float b), bf16(lo float a)} by
// gathering the top 2 bytes of each source (truncation: exact for 0/1 adjacency).
static __device__ __forceinline__ uint32_t pk_trunc(float lo, float hi) {
    return __builtin_amdgcn_perm(__float_as_uint(hi), __float_as_uint(lo), 0x07060302u);
}

// ---------------------------------------------------------------------------
// Kernel 1: per-node MLP score  e_j = relu(h_j @ W1 + b1) @ W2 + b2
// one 64-thread block per node; h row staged in LDS
// ---------------------------------------------------------------------------
__global__ void __launch_bounds__(64)
k_scores(const float* __restrict__ h, const float* __restrict__ W1,
         const float* __restrict__ b1, const float* __restrict__ W2,
         const float* __restrict__ b2, float* __restrict__ e) {
    __shared__ float hrow[DIMS];
    __shared__ float red[HID];
    const int j = blockIdx.x;
    const int t = threadIdx.x;            // 0..63 = hidden unit
    hrow[t]      = h[j * DIMS + t];
    hrow[t + 64] = h[j * DIMS + t + 64];
    __syncthreads();
    float acc = b1[t];
#pragma unroll 8
    for (int d = 0; d < DIMS; ++d)
        acc = fmaf(hrow[d], W1[d * HID + t], acc);   // W1[d][t]: coalesced over t
    acc = fmaxf(acc, 0.0f);
    red[t] = acc * W2[t];
    __syncthreads();
    for (int s = HID / 2; s > 0; s >>= 1) {
        if (t < s) red[t] += red[t + s];
        __syncthreads();
    }
    if (t == 0) e[j] = red[0] + b2[0];
}

// ---------------------------------------------------------------------------
// Kernel 2: global max of e (single block)
// ---------------------------------------------------------------------------
__global__ void __launch_bounds__(256)
k_gmax(const float* __restrict__ e, float* __restrict__ gmax) {
    __shared__ float red[256];
    const int t = threadIdx.x;
    float m = -INFINITY;
    for (int i = t; i < N_NODES; i += 256) m = fmaxf(m, e[i]);
    red[t] = m;
    __syncthreads();
    for (int s = 128; s > 0; s >>= 1) {
        if (t < s) red[t] = fmaxf(red[t], red[t + s]);
        __syncthreads();
    }
    if (t == 0) gmax[0] = red[0];
}

// ---------------------------------------------------------------------------
// Kernel 3: build B matrix in WMMA B-fragment layout (bf16, RNE)
//   logical B[k][n]:  n<128 -> w_k * h[k][n],  n==128 -> w_k,  n==129 -> 1,  else 0
//   fragment addr:  halfs[ ((k/32)*9 + n/16)*512 + ((n&15) + 16*(k%32>=16))*16 + (k%32)&15 ]
// ---------------------------------------------------------------------------
__global__ void __launch_bounds__(256)
k_buildB(const float* __restrict__ h, const float* __restrict__ e,
         const float* __restrict__ gmax, uint16_t* __restrict__ B) {
    const int idx = blockIdx.x * 256 + threadIdx.x;     // < 8192*144
    const int k = idx / NCOLS;
    const int n = idx - k * NCOLS;
    const float w = expf(e[k] - gmax[0]);
    float val;
    if (n < DIMS)            val = w * h[k * DIMS + n];
    else if (n == DIMS)      val = w;            // softmax denominator column
    else if (n == DIMS + 1)  val = 1.0f;         // degree column
    else                     val = 0.0f;         // pad
    uint32_t ub = __float_as_uint(val);
    ub += 0x7FFFu + ((ub >> 16) & 1u);           // round-to-nearest-even to bf16
    const int kt = k >> 5, r = k & 31;
    const int off = (((kt * NT) + (n >> 4)) * 32 + (n & 15) + ((r >= 16) ? 16 : 0)) * 16 + (r & 15);
    B[off] = (uint16_t)(ub >> 16);
}

// ---------------------------------------------------------------------------
// Kernel 4: zero the f32 accumulation targets (d_out and sn)
// ---------------------------------------------------------------------------
__global__ void __launch_bounds__(256)
k_zero(float* __restrict__ out, float* __restrict__ sn) {
    const int idx = blockIdx.x * 256 + threadIdx.x;     // < 1048576 + 131072
    if (idx < N_NODES * DIMS) out[idx] = 0.0f;
    else                      sn[idx - N_NODES * DIMS] = 0.0f;
}

// ---------------------------------------------------------------------------
// Kernel 5: the GEMM  A[8192x8192](0/1 fp32 -> bf16) @ B[8192x144](bf16) -> f32
//   grid = 64 M-blocks x 4 K-chunks; block = 256 threads = 8 waves x 16 rows
//   per wave per K-step: batch-issue 4 A loads + 18 B loads (one clause, one
//   wait), pack A via v_perm_b32, then 9 back-to-back v_wmma_f32_16x16x32_bf16
// ---------------------------------------------------------------------------
__global__ void __launch_bounds__(256)
k_gemm(const float* __restrict__ A, const uint16_t* __restrict__ B,
       float* __restrict__ out, float* __restrict__ sn) {
    const int lane   = threadIdx.x & 31;
    const int wave   = threadIdx.x >> 5;
    const int mblk   = blockIdx.x & 63;
    const int kchunk = blockIdx.x >> 6;

    const int mbase = mblk * MTILE + wave * 16;
    const int row   = mbase + (lane & 15);
    const float* Arow = A + (size_t)row * N_NODES;
    const int koff  = (lane < 16) ? 0 : 16;    // A frag: lane<16 -> K 0-15, lane>=16 -> K 16-31
    const int kbase = kchunk * KCHUNK;

    v8f c[NT] = {};

#pragma unroll 2
    for (int kk = 0; kk < KCHUNK / 32; ++kk) {
        const int k0 = kbase + kk * 32;

        // --- batch-issue ALL loads for this K-step (A first, then B) ---
        const float4* ap  = (const float4*)(Arow + k0 + koff);
        const float4* ap2 = (const float4*)(Arow + k0 + 8 + koff);
        float4 f0 = ap[0];
        float4 f1 = ap[1];
        float4 f2 = ap2[0];
        float4 f3 = ap2[1];

        const uint16_t* bbase = B + (size_t)(k0 >> 5) * (NT * 512) + lane * 16;
        v16bf bv[NT];
#pragma unroll
        for (int nt = 0; nt < NT; ++nt)
            bv[nt] = *(const v16bf*)(bbase + nt * 512);

        __builtin_prefetch(Arow + k0 + 96, 0, 1);   // global_prefetch_b8, 3 iters ahead

        // --- pack A to bf16 fragment (v_perm_b32, 8 VALU) ---
        v8u pkd;
        pkd[0] = pk_trunc(f0.x, f0.y);
        pkd[1] = pk_trunc(f0.z, f0.w);
        pkd[2] = pk_trunc(f1.x, f1.y);
        pkd[3] = pk_trunc(f1.z, f1.w);
        pkd[4] = pk_trunc(f2.x, f2.y);
        pkd[5] = pk_trunc(f2.z, f2.w);
        pkd[6] = pk_trunc(f3.x, f3.y);
        pkd[7] = pk_trunc(f3.z, f3.w);
        const v16bf av = __builtin_bit_cast(v16bf, pkd);

        // --- 9 WMMAs back-to-back ---
#pragma unroll
        for (int nt = 0; nt < NT; ++nt)
            c[nt] = __builtin_amdgcn_wmma_f32_16x16x32_bf16(
                false, av, false, bv[nt], (short)0, c[nt], false, false);
    }

    // C/D layout: VGPR v, lane l -> M = v + (l<16?0:8), N = l%16
    const int ncol = lane & 15;
    const int mofs = (lane < 16) ? 0 : 8;
#pragma unroll
    for (int v = 0; v < 8; ++v) {
        const int rr = mbase + v + mofs;
#pragma unroll
        for (int nt = 0; nt < 8; ++nt) {
            __hip_atomic_fetch_add(&out[rr * DIMS + nt * 16 + ncol], c[nt][v],
                                   __ATOMIC_RELAXED, __HIP_MEMORY_SCOPE_AGENT);
        }
        __hip_atomic_fetch_add(&sn[rr * 16 + ncol], c[8][v],
                               __ATOMIC_RELAXED, __HIP_MEMORY_SCOPE_AGENT);
    }
}

// ---------------------------------------------------------------------------
// Kernel 6: out[i,c] *= num_i / s_i   (0 for isolated nodes, matching reference)
// ---------------------------------------------------------------------------
__global__ void __launch_bounds__(256)
k_scale(float* __restrict__ out, const float* __restrict__ sn) {
    const int idx = blockIdx.x * 256 + threadIdx.x;     // < 8192*128
    const int i = idx >> 7;
    const float s   = sn[i * 16 + 0];
    const float num = sn[i * 16 + 1];
    const float sc = (s > 0.0f) ? (num / s) : 0.0f;
    out[idx] *= sc;
}

// ---------------------------------------------------------------------------
extern "C" void kernel_launch(void* const* d_in, const int* in_sizes, int n_in,
                              void* d_out, int out_size, void* d_ws, size_t ws_size,
                              hipStream_t stream) {
    (void)in_sizes; (void)n_in; (void)out_size; (void)ws_size;
    const float* A  = (const float*)d_in[0];   // graph_info [8192,8192]
    const float* h  = (const float*)d_in[1];   // [8192,128]
    const float* W1 = (const float*)d_in[2];   // [128,64]
    const float* b1 = (const float*)d_in[3];   // [64]
    const float* W2 = (const float*)d_in[4];   // [64,1]
    const float* b2 = (const float*)d_in[5];   // [1]
    float* out = (float*)d_out;

    uint8_t* ws = (uint8_t*)d_ws;
    float*    e    = (float*)(ws + WS_E);
    float*    gmax = (float*)(ws + WS_GMAX);
    uint16_t* Bws  = (uint16_t*)(ws + WS_B);
    float*    sn   = (float*)(ws + WS_SN);

    k_scores<<<dim3(N_NODES), dim3(64), 0, stream>>>(h, W1, b1, W2, b2, e);
    k_gmax  <<<dim3(1), dim3(256), 0, stream>>>(e, gmax);
    k_buildB<<<dim3((N_NODES * NCOLS) / 256), dim3(256), 0, stream>>>(h, e, gmax, Bws);
    k_zero  <<<dim3((N_NODES * DIMS + N_NODES * 16) / 256), dim3(256), 0, stream>>>(out, sn);
    k_gemm  <<<dim3(64 * KSPLIT), dim3(256), 0, stream>>>(A, Bws, out, sn);
    k_scale <<<dim3((N_NODES * DIMS) / 256), dim3(256), 0, stream>>>(out, sn);
}
